// TopkAttention_403726925850
// MI455X (gfx1250) — compile-verified, matrix operand order checked by reading, not run on hardware
//
#include <hip/hip_runtime.h>
#include <math.h>

#define BB 1
#define NQ 512
#define HID 4096
#define NH 32
#define KVH 8
#define DD 128
#define NPRE 4096
#define TOPK 64

typedef __attribute__((ext_vector_type(16))) __bf16 v16bf;
typedef __attribute__((ext_vector_type(8)))  float  v8f;

union FragU { v16bf v; uint4 u[2]; };

// Load one 16x32 bf16 WMMA fragment (A layout; B uses same mapping on a
// pre-transposed [N][K] matrix). `rowk` points at element (row, kbase) where
// row = tile_base + (lane & 15). Per ISA: element i of lane L holds
// K = kbase + (i/8)*16 + (L/16)*8 + (i%8)  -> two contiguous 16B chunks.
__device__ __forceinline__ v16bf load_frag(const __bf16* rowk, int lane) {
  const __bf16* p = rowk + ((lane >> 4) << 3);
  FragU f;
  f.u[0] = *reinterpret_cast<const uint4*>(p);
  f.u[1] = *reinterpret_cast<const uint4*>(p + 16);
  return f.v;
}

// ---------------- elementwise convert / transpose ----------------
__global__ void cvt_f32_bf16(const float* __restrict__ x, __bf16* __restrict__ y, long n) {
  long i = (long)blockIdx.x * blockDim.x + threadIdx.x;
  if (i < n) y[i] = (__bf16)x[i];
}

// x is [K][N] row-major f32 ; y is [N][K] row-major bf16 (transposed)
__global__ void cvt_t_f32_bf16(const float* __restrict__ x, __bf16* __restrict__ y, int K, int N) {
  long i = (long)blockIdx.x * blockDim.x + threadIdx.x;
  if (i < (long)K * N) {
    int k = (int)(i / N);
    int n = (int)(i % N);
    y[(long)n * K + k] = (__bf16)x[i];
  }
}

// ------- register-blocked WMMA bf16 GEMM: C[M][N] = A[M][K] * Bt[N][K]^T ----
// One wave computes a 32x32 C tile (2x2 of 16x16): each k-step loads 2 A frags
// + 2 B frags and issues 4 WMMAs -> 2x fragment reuse vs the naive tile.
__global__ __launch_bounds__(32) void gemm_bf16(const __bf16* __restrict__ A,
                                                const __bf16* __restrict__ Bt,
                                                float* __restrict__ C,
                                                int M, int N, int K) {
  const int lane = threadIdx.x;
  const int col = lane & 15;
  const long nb = (long)blockIdx.x * 32;
  const long mb = (long)blockIdx.y * 32;
  const __bf16* arow0 = A  + (mb + col)      * (long)K;
  const __bf16* arow1 = A  + (mb + 16 + col) * (long)K;
  const __bf16* brow0 = Bt + (nb + col)      * (long)K;
  const __bf16* brow1 = Bt + (nb + 16 + col) * (long)K;
  v8f acc00 = {}, acc01 = {}, acc10 = {}, acc11 = {};
  for (int k = 0; k < K; k += 32) {
    if (k + 256 < K) {  // pull next chunks toward the WGP (global_prefetch_b8)
      __builtin_prefetch(arow0 + k + 256, 0, 1);
      __builtin_prefetch(arow1 + k + 256, 0, 1);
      __builtin_prefetch(brow0 + k + 256, 0, 1);
      __builtin_prefetch(brow1 + k + 256, 0, 1);
    }
    v16bf a0 = load_frag(arow0 + k, lane);
    v16bf a1 = load_frag(arow1 + k, lane);
    v16bf b0 = load_frag(brow0 + k, lane);
    v16bf b1 = load_frag(brow1 + k, lane);
    acc00 = __builtin_amdgcn_wmma_f32_16x16x32_bf16(false, a0, false, b0, (short)0, acc00, false, false);
    acc01 = __builtin_amdgcn_wmma_f32_16x16x32_bf16(false, a0, false, b1, (short)0, acc01, false, false);
    acc10 = __builtin_amdgcn_wmma_f32_16x16x32_bf16(false, a1, false, b0, (short)0, acc10, false, false);
    acc11 = __builtin_amdgcn_wmma_f32_16x16x32_bf16(false, a1, false, b1, (short)0, acc11, false, false);
  }
  const long crow = mb + ((lane >> 4) << 3);
  float* c00 = C + crow * N + nb + col;
#pragma unroll
  for (int r = 0; r < 8; ++r) {
    c00[(long)r * N]             = acc00[r];
    c00[(long)r * N + 16]        = acc01[r];
    c00[(long)(r + 16) * N]      = acc10[r];
    c00[(long)(r + 16) * N + 16] = acc11[r];
  }
}

// ---------------- RoPE (scale folded into q) ----------------
__global__ void rope_q(const float* __restrict__ Q, const float* __restrict__ cb,
                       const float* __restrict__ sb, __bf16* __restrict__ qout) {
  long i = (long)blockIdx.x * blockDim.x + threadIdx.x;
  if (i >= (long)NQ * NH * DD) return;
  int d  = (int)(i % DD);
  long t = i / DD;
  int h  = (int)(t % NH);
  int qi = (int)(t / NH);
  const float* row = Q + (long)qi * (NH * DD) + (long)h * DD;
  float x = row[d];
  float o = row[(d < DD / 2) ? d + DD / 2 : d - DD / 2];
  float c = cb[qi * DD + d], s = sb[qi * DD + d];
  float r = (d < DD / 2) ? (x * c - o * s) : (x * c + o * s);
  r *= 0.08838834764831845f; // 1/sqrt(128)
  qout[((long)h * NQ + qi) * DD + d] = (__bf16)r;
}

__global__ void rope_kv(const float* __restrict__ Kf, const float* __restrict__ Vf,
                        const float* __restrict__ cb, const float* __restrict__ sb,
                        __bf16* __restrict__ kout, __bf16* __restrict__ vout,
                        __bf16* __restrict__ vtout) {
  long i = (long)blockIdx.x * blockDim.x + threadIdx.x;
  if (i >= (long)NQ * KVH * DD) return;
  int d  = (int)(i % DD);
  long t = i / DD;
  int h  = (int)(t % KVH);
  int qi = (int)(t / KVH);
  const float* krow = Kf + (long)qi * (KVH * DD) + (long)h * DD;
  float x = krow[d];
  float o = krow[(d < DD / 2) ? d + DD / 2 : d - DD / 2];
  float c = cb[qi * DD + d], s = sb[qi * DD + d];
  float r = (d < DD / 2) ? (x * c - o * s) : (x * c + o * s);
  kout[((long)h * NQ + qi) * DD + d] = (__bf16)r;
  float v = Vf[(long)qi * (KVH * DD) + (long)h * DD + d];
  vout[((long)h * NQ + qi) * DD + d] = (__bf16)v;
  vtout[((long)h * DD + d) * NQ + qi] = (__bf16)v;
}

// ---------------- top-64 per row via LDS iterative argmax ----------------
__global__ __launch_bounds__(128) void topk_rows(const float* __restrict__ S,
                                                 float* __restrict__ vals,
                                                 int* __restrict__ idxs) {
  __shared__ float sv[NPRE];
  __shared__ float rv[128];
  __shared__ int   ri[128];
  const int row = blockIdx.x, tid = threadIdx.x;
  const float* srow = S + (long)row * NPRE;
  for (int j = tid; j < NPRE; j += 128) sv[j] = srow[j];
  __syncthreads();
  for (int t = 0; t < TOPK; ++t) {
    float bv = -INFINITY; int bi = 0;
    for (int j = tid; j < NPRE; j += 128) {
      float v = sv[j];
      if (v > bv) { bv = v; bi = j; }
    }
    rv[tid] = bv; ri[tid] = bi;
    __syncthreads();
    for (int s = 64; s > 0; s >>= 1) {
      if (tid < s && rv[tid + s] > rv[tid]) { rv[tid] = rv[tid + s]; ri[tid] = ri[tid + s]; }
      __syncthreads();
    }
    if (tid == 0) {
      vals[(long)row * TOPK + t] = rv[0];
      idxs[(long)row * TOPK + t] = ri[0];
      sv[ri[0]] = -INFINITY;
    }
    __syncthreads();
  }
}

// ---------------- top-k gather: xtop = sum exp(v_t) * pv[idx_t], tsum = sum exp
__global__ __launch_bounds__(32) void topk_gather(const float* __restrict__ vals,
                                                  const int* __restrict__ idxs,
                                                  const float* __restrict__ pv,
                                                  float* __restrict__ xtop,
                                                  float* __restrict__ tsum) {
  const int qi = blockIdx.x, h = blockIdx.y, lane = threadIdx.x;
  const int kvh = h / (NH / KVH);
  const float* vrow = vals + ((long)h * NQ + qi) * TOPK;
  const int*   irow = idxs + ((long)h * NQ + qi) * TOPK;
  const int d = lane * 4;
  float x0 = 0.f, x1 = 0.f, x2 = 0.f, x3 = 0.f, s = 0.f;
  for (int t = 0; t < TOPK; ++t) {
    float w = __expf(vrow[t]);
    s += w;
    const float* p = pv + ((long)kvh * NPRE + irow[t]) * DD + d;
    x0 += w * p[0]; x1 += w * p[1]; x2 += w * p[2]; x3 += w * p[3];
  }
  float* xo = xtop + ((long)h * NQ + qi) * DD + d;
  xo[0] = x0; xo[1] = x1; xo[2] = x2; xo[3] = x3;
  if (lane == 0) tsum[(long)h * NQ + qi] = s;
}

// ---------------- fused causal dense attention + top-k merge -------------
// one wave per (16-query tile, head); exp without max matches reference
__global__ __launch_bounds__(32) void dense_attn(const __bf16* __restrict__ qb,  // [NH][NQ][DD]
                                                 const __bf16* __restrict__ kb,  // [KVH][NQ][DD]
                                                 const __bf16* __restrict__ vt,  // [KVH][DD][NQ]
                                                 const float* __restrict__ xtop, // [NH][NQ][DD]
                                                 const float* __restrict__ tsum, // [NH][NQ]
                                                 __bf16* __restrict__ attout) {  // [NQ][NH*DD]
  __shared__ __bf16 Pl[16 * 32];
  const int qt = blockIdx.x, h = blockIdx.y, lane = threadIdx.x;
  const int kvh = h / (NH / KVH);
  const int qbase = qt * 16;
  const int grp = lane >> 4, col = lane & 15;

  const __bf16* qrow = qb + ((long)h * NQ + qbase + col) * DD;
  v16bf qf[4];
#pragma unroll
  for (int c = 0; c < 4; ++c) qf[c] = load_frag(qrow + c * 32, lane);

  v8f acc[8] = {};
  float rs[8] = {0.f, 0.f, 0.f, 0.f, 0.f, 0.f, 0.f, 0.f};

  const int kend = qbase + 16; // exclusive causal bound
  for (int kbase = 0; kbase < kend; kbase += 32) {
#pragma unroll
    for (int s = 0; s < 2; ++s) {
      const int kb0 = kbase + s * 16;
      const __bf16* krow = kb + ((long)kvh * NQ + kb0 + col) * DD;
      v8f S = {};
#pragma unroll
      for (int c = 0; c < 4; ++c) {
        v16bf bf = load_frag(krow + c * 32, lane);
        S = __builtin_amdgcn_wmma_f32_16x16x32_bf16(false, qf[c], false, bf, (short)0, S, false, false);
      }
      const int n = kb0 + col;
#pragma unroll
      for (int r = 0; r < 8; ++r) {
        const int m = qbase + r + 8 * grp;
        float p = (n <= m) ? __expf(S[r]) : 0.0f;
        rs[r] += p;
        Pl[(r + 8 * grp) * 32 + s * 16 + col] = (__bf16)p;
      }
    }
    __syncthreads();
    v16bf pf = load_frag(&Pl[col * 32], lane);
#pragma unroll
    for (int nt = 0; nt < 8; ++nt) {
      const __bf16* vrow = vt + ((long)kvh * DD + nt * 16 + col) * NQ + kbase;
      v16bf bf = load_frag(vrow, lane);
      acc[nt] = __builtin_amdgcn_wmma_f32_16x16x32_bf16(false, pf, false, bf, (short)0, acc[nt], false, false);
    }
    __syncthreads();
  }

#pragma unroll
  for (int r = 0; r < 8; ++r) {
    float v = rs[r];
    for (int off = 1; off < 16; off <<= 1) v += __shfl_xor(v, off, 32);
    rs[r] = v;
  }
#pragma unroll
  for (int r = 0; r < 8; ++r) {
    const int m = qbase + r + 8 * grp;
    const float denom = rs[r] + tsum[(long)h * NQ + m];
#pragma unroll
    for (int nt = 0; nt < 8; ++nt) {
      const int d = nt * 16 + col;
      float val = (acc[nt][r] + xtop[((long)h * NQ + m) * DD + d]) / denom;
      attout[(long)m * (NH * DD) + (long)h * DD + d] = (__bf16)val;
    }
  }
}

extern "C" void kernel_launch(void* const* d_in, const int* in_sizes, int n_in,
                              void* d_out, int out_size, void* d_ws, size_t ws_size,
                              hipStream_t stream) {
  (void)in_sizes; (void)n_in; (void)out_size; (void)ws_size;
  const float* hs   = (const float*)d_in[0];
  const float* cosb = (const float*)d_in[1];
  const float* sinb = (const float*)d_in[2];
  const float* pk   = (const float*)d_in[3];
  const float* pv   = (const float*)d_in[4];
  const float* Wq   = (const float*)d_in[5];
  const float* Wk   = (const float*)d_in[6];
  const float* Wv   = (const float*)d_in[7];
  const float* Wo   = (const float*)d_in[8];
  float* out = (float*)d_out;

  char* ws = (char*)d_ws;
  size_t off = 0;
  auto alloc = [&](size_t bytes) -> void* {
    void* p = ws + off;
    off = (off + bytes + 255) & ~(size_t)255;
    return p;
  };

  __bf16* hsb  = (__bf16*)alloc((size_t)NQ * HID * 2);
  __bf16* wqt  = (__bf16*)alloc((size_t)HID * HID * 2);
  __bf16* wkt  = (__bf16*)alloc((size_t)(KVH * DD) * HID * 2);
  __bf16* wvt  = (__bf16*)alloc((size_t)(KVH * DD) * HID * 2);
  __bf16* wot  = (__bf16*)alloc((size_t)HID * (NH * DD) * 2);
  __bf16* pkb  = (__bf16*)alloc((size_t)KVH * NPRE * DD * 2);
  float*  Qf   = (float*) alloc((size_t)NQ * NH * DD * 4);
  float*  Kf   = (float*) alloc((size_t)NQ * KVH * DD * 4);
  float*  Vf   = (float*) alloc((size_t)NQ * KVH * DD * 4);
  __bf16* qb   = (__bf16*)alloc((size_t)NH * NQ * DD * 2);
  __bf16* kb2  = (__bf16*)alloc((size_t)KVH * NQ * DD * 2);
  __bf16* vb   = (__bf16*)alloc((size_t)KVH * NQ * DD * 2);
  __bf16* vtb  = (__bf16*)alloc((size_t)KVH * DD * NQ * 2);
  float*  scores = (float*)alloc((size_t)NQ * NPRE * 4);
  float*  tvals  = (float*)alloc((size_t)NH * NQ * TOPK * 4);
  int*    tidx   = (int*)  alloc((size_t)NH * NQ * TOPK * 4);
  float*  tsum   = (float*)alloc((size_t)NH * NQ * 4);
  float*  xtop   = (float*)alloc((size_t)NH * NQ * DD * 4);
  __bf16* attout = (__bf16*)alloc((size_t)NQ * NH * DD * 2);

  const int T = 256;
  long n;

  n = (long)NQ * HID;
  cvt_f32_bf16<<<dim3((unsigned)((n + T - 1) / T)), T, 0, stream>>>(hs, hsb, n);
  n = (long)HID * (NH * DD);
  cvt_t_f32_bf16<<<dim3((unsigned)((n + T - 1) / T)), T, 0, stream>>>(Wq, wqt, HID, NH * DD);
  n = (long)HID * (KVH * DD);
  cvt_t_f32_bf16<<<dim3((unsigned)((n + T - 1) / T)), T, 0, stream>>>(Wk, wkt, HID, KVH * DD);
  cvt_t_f32_bf16<<<dim3((unsigned)((n + T - 1) / T)), T, 0, stream>>>(Wv, wvt, HID, KVH * DD);
  n = (long)(NH * DD) * HID;
  cvt_t_f32_bf16<<<dim3((unsigned)((n + T - 1) / T)), T, 0, stream>>>(Wo, wot, NH * DD, HID);
  n = (long)KVH * NPRE * DD;
  cvt_f32_bf16<<<dim3((unsigned)((n + T - 1) / T)), T, 0, stream>>>(pk, pkb, n);

  // Q/K/V projections (32x32 C tile per wave)
  gemm_bf16<<<dim3((NH * DD) / 32, NQ / 32), 32, 0, stream>>>(hsb, wqt, Qf, NQ, NH * DD, HID);
  gemm_bf16<<<dim3((KVH * DD) / 32, NQ / 32), 32, 0, stream>>>(hsb, wkt, Kf, NQ, KVH * DD, HID);
  gemm_bf16<<<dim3((KVH * DD) / 32, NQ / 32), 32, 0, stream>>>(hsb, wvt, Vf, NQ, KVH * DD, HID);

  n = (long)NQ * NH * DD;
  rope_q<<<dim3((unsigned)((n + T - 1) / T)), T, 0, stream>>>(Qf, cosb, sinb, qb);
  n = (long)NQ * KVH * DD;
  rope_kv<<<dim3((unsigned)((n + T - 1) / T)), T, 0, stream>>>(Kf, Vf, cosb, sinb, kb2, vb, vtb);

  // Per-head prefix scores + top-64 (score buffer reused; stream serializes)
  for (int h = 0; h < NH; ++h) {
    const __bf16* qh  = qb  + (size_t)h * NQ * DD;
    const __bf16* pkh = pkb + (size_t)(h / (NH / KVH)) * NPRE * DD;
    gemm_bf16<<<dim3(NPRE / 32, NQ / 32), 32, 0, stream>>>(qh, pkh, scores, NQ, NPRE, DD);
    topk_rows<<<dim3(NQ), 128, 0, stream>>>(scores,
                                            tvals + (size_t)h * NQ * TOPK,
                                            tidx + (size_t)h * NQ * TOPK);
  }

  topk_gather<<<dim3(NQ, NH), 32, 0, stream>>>(tvals, tidx, pv, xtop, tsum);
  dense_attn<<<dim3(NQ / 16, NH), 32, 0, stream>>>(qb, kb2, vtb, xtop, tsum, attout);

  // output projection straight into d_out (f32)
  gemm_bf16<<<dim3(HID / 32, NQ / 32), 32, 0, stream>>>(attout, wot, out, NQ, HID, NH * DD);
}
